// GGNN_13580686590233
// MI455X (gfx1250) — matrix operand order repaired and checked by reading, not compiled, not run on hardware
//
#include <hip/hip_runtime.h>

// ---------------------------------------------------------------------------
// GGNN on MI455X (gfx1250): bf16 WMMA GEMMs for edge-message and GRU matmuls.
// B=32 N=512 H=128 E=131072 T=9, 8 propagate steps, dims per layer {128,256,128,384}.
// Heavy math: v_wmma_f32_16x16x32_bf16 (fp32 accumulate).
// A-tile staging: global_load_async_to_lds_b128 (ASYNCcnt) when available.
// ---------------------------------------------------------------------------

typedef __bf16 bf16;
typedef __attribute__((ext_vector_type(16))) __bf16 bf16x16;
typedef __attribute__((ext_vector_type(8)))  float  floatx8;
typedef __attribute__((ext_vector_type(4)))  int    v4i;

static constexpr int Bc  = 32;
static constexpr int Nc  = 512;
static constexpr int Hc  = 128;
static constexpr int Ec  = 131072;
static constexpr int Tc  = 9;
static constexpr int BNc = Bc * Nc;          // 16384
static constexpr int G3H = 3 * Hc;           // 384

struct Meta {
    int counts[16];
    int seg_off[16];
    int tile_off[16];
    int cursor[16];
    int total_tiles;
    int pad[15];
};

union V16 { bf16x16 v; uint4 q[2]; };

__device__ __forceinline__ floatx8 wmma_bf16(bf16x16 a, bf16x16 b, floatx8 c) {
    return __builtin_amdgcn_wmma_f32_16x16x32_bf16(false, a, false, b, (short)0, c,
                                                   false, false);
}

// ---- async global->LDS 16B copy (gfx1250 ASYNCcnt path) with safe fallback ----
#if defined(__has_builtin)
#if __has_builtin(__builtin_amdgcn_global_load_async_to_lds_b128)
#define HAVE_ASYNC_LDS 1
#endif
#endif
#ifndef HAVE_ASYNC_LDS
#define HAVE_ASYNC_LDS 0
#endif

typedef __attribute__((address_space(1))) v4i* gptr_v4i;
typedef __attribute__((address_space(3))) v4i* lptr_v4i;

__device__ __forceinline__ void async_copy16(const bf16* g, bf16* l) {
#if HAVE_ASYNC_LDS
    __builtin_amdgcn_global_load_async_to_lds_b128(
        (gptr_v4i)(__UINTPTR_TYPE__)g, (lptr_v4i)(__UINTPTR_TYPE__)l, 0, 0);
#else
    *(uint4*)l = *(const uint4*)g;
#endif
}

__device__ __forceinline__ void async_wait() {
#if HAVE_ASYNC_LDS
#if __has_builtin(__builtin_amdgcn_s_wait_asynccnt)
    __builtin_amdgcn_s_wait_asynccnt(0);
#else
    asm volatile("s_wait_asynccnt 0x0" ::: "memory");
#endif
#endif
}

// ------------------------------- prep kernels ------------------------------

__global__ void k_hist(const int* __restrict__ eids, Meta* __restrict__ m) {
    int e = blockIdx.x * blockDim.x + threadIdx.x;
    if (e < Ec) atomicAdd(&m->counts[eids[e * 4 + 0]], 1);
}

__global__ void k_scan(Meta* __restrict__ m) {
    if (blockIdx.x == 0 && threadIdx.x == 0) {
        int off = 0, toff = 0;
        for (int t = 0; t < Tc; ++t) {
            m->seg_off[t]  = off;
            m->cursor[t]   = off;
            m->tile_off[t] = toff;
            toff += (m->counts[t] + 63) / 64;
            off  += m->counts[t];
        }
        m->seg_off[Tc]  = off;
        m->tile_off[Tc] = toff;
        m->total_tiles  = toff;
    }
}

__global__ void k_scatter(const int* __restrict__ eids, Meta* __restrict__ m,
                          int* __restrict__ order) {
    int e = blockIdx.x * blockDim.x + threadIdx.x;
    if (e < Ec) {
        int t = eids[e * 4 + 0];
        int p = atomicAdd(&m->cursor[t], 1);
        order[p] = e;
    }
}

// dst[b][n][k] = (bf16) src[b][k][n]   (row-major src K x Ncols per batch)
__global__ void k_transpose_bf16(bf16* __restrict__ dst, const float* __restrict__ src,
                                 int batch, int K, int Ncols) {
    long total = (long)batch * K * Ncols;
    for (long idx = blockIdx.x * (long)blockDim.x + threadIdx.x; idx < total;
         idx += (long)gridDim.x * blockDim.x) {
        long b   = idx / ((long)K * Ncols);
        long rem = idx - b * (long)K * Ncols;
        long n   = rem / K;
        long k   = rem - n * K;
        dst[idx] = (bf16)src[b * (long)K * Ncols + k * Ncols + n];
    }
}

__global__ void k_copy_f32(float* __restrict__ dst, const float* __restrict__ src, int n) {
    int i = blockIdx.x * blockDim.x + threadIdx.x;
    if (i < n) dst[i] = src[i];
}

__global__ void k_f32_to_bf16(bf16* __restrict__ dst, const float* __restrict__ src, int n) {
    int i = blockIdx.x * blockDim.x + threadIdx.x;
    if (i < n) dst[i] = (bf16)src[i];
}

// x = concat(p0, p1, p2) row-wise (each part is BN x 128 fp32), output bf16 BN x din
__global__ void k_concat_bf16(bf16* __restrict__ dst,
                              const float* __restrict__ p0, const float* __restrict__ p1,
                              const float* __restrict__ p2, int n0, int n1, int n2) {
    int din = n0 + n1 + n2;
    long total = (long)BNc * din;
    for (long idx = blockIdx.x * (long)blockDim.x + threadIdx.x; idx < total;
         idx += (long)gridDim.x * blockDim.x) {
        long row = idx / din;
        int  c   = (int)(idx - row * din);
        float v;
        if (c < n0)           v = p0[row * Hc + c];
        else if (c < n0 + n1) v = p1[row * Hc + (c - n0)];
        else                  v = p2[row * Hc + (c - n0 - n1)];
        dst[idx] = (bf16)v;
    }
}

// ------------------------ edge gather->GEMM->scatter -----------------------
// Block: 1024 thr = 32 waves. 64 edges (one type) x 128 cols. K = 128, bf16 WMMA.
// Source rows gathered from the bf16 state copy via async-to-LDS.

__global__ __launch_bounds__(1024, 1) void k_edge_gemm(
    const bf16* __restrict__ hbf, const int* __restrict__ eids,
    const int* __restrict__ order, const Meta* __restrict__ m,
    const bf16* __restrict__ WtType,  // [4*T][n=128][k=128] bf16
    const float* __restrict__ biasT,  // [4][T][128] fp32
    float* __restrict__ agg, int layer) {
    constexpr int AS = 136;  // padded LDS stride (bf16) to spread banks
    __shared__ __align__(16) bf16 As[64 * AS];
    __shared__ int eidxS[64];
    __shared__ int srcRowS[64];
    __shared__ int tgtS[64];

    int tile = blockIdx.x;
    if (tile >= m->total_tiles) return;
    int t = 0;
    while (tile >= m->tile_off[t + 1]) ++t;
    int local    = tile - m->tile_off[t];
    int segStart = m->seg_off[t];
    int cnt      = m->counts[t];
    int base     = segStart + local * 64;
    int tid      = threadIdx.x;

    if (tid < 64) {
        int slot = base + tid;
        if (slot < segStart + cnt) {
            int e = order[slot];
            eidxS[tid]   = e;
            srcRowS[tid] = eids[e * 4 + 1] * Nc + eids[e * 4 + 2];
            tgtS[tid]    = eids[e * 4 + 1] * Nc + eids[e * 4 + 3];
        } else {
            // padding row: gather node 0 (its D-rows are skipped in the epilogue)
            eidxS[tid] = -1; srcRowS[tid] = 0; tgtS[tid] = 0;
        }
    }
    __syncthreads();

    {   // stage 64 rows x 128 bf16 (256B/row): 16 lanes/row, 16B per lane, async
        int r = tid >> 4;          // 0..63
        int c = (tid & 15) * 8;    // bf16 column of this 16B chunk
        async_copy16(hbf + (long)srcRowS[r] * Hc + c, &As[r * AS + c]);
    }
    async_wait();
    __syncthreads();

    int w    = tid >> 5;        // wave 0..31
    int lane = tid & 31;
    int esub = w >> 3;          // 16-edge subtile 0..3
    int ct   = w & 7;           // col tile 0..7
    int nloc = ct * 16 + (lane & 15);
    int half = lane >> 4;
    int mrow16 = lane & 15;

    const bf16* Wt = WtType + ((long)(layer * Tc + t) * Hc + nloc) * Hc;  // row n, K contiguous
    __builtin_prefetch(Wt, 0, 3);  // global_prefetch_b8

    floatx8 acc = {0.f, 0.f, 0.f, 0.f, 0.f, 0.f, 0.f, 0.f};
#pragma unroll
    for (int kb = 0; kb < Hc; kb += 32) {
        V16 a, b;
        const bf16* arow = &As[(esub * 16 + mrow16) * AS + kb + 8 * half];
        a.q[0] = *(const uint4*)(arow);
        a.q[1] = *(const uint4*)(arow + 16);
        const bf16* brow = Wt + kb + 8 * half;
        b.q[0] = *(const uint4*)(brow);
        b.q[1] = *(const uint4*)(brow + 16);
        acc = wmma_bf16(a.v, b.v, acc);
    }

    float bias = biasT[((long)layer * Tc + t) * Hc + nloc];
#pragma unroll
    for (int v = 0; v < 8; ++v) {
        int mr = v + 8 * half;
        int r  = esub * 16 + mr;
        if (eidxS[r] >= 0)
            atomicAdd(&agg[(long)tgtS[r] * Hc + nloc], acc[v] + bias);
    }
}

// --------------------------------- GRU -------------------------------------
// Block: 768 thr = 24 waves; 16 nodes x 384 gate cols. Two WMMA chains
// (x@W over K=DIN, h@U over K=128), fused sigmoid/tanh gates in LDS.

template <int DIN>
__global__ __launch_bounds__(768, 1) void k_gru(
    const bf16* __restrict__ xbf,   // [BN][DIN]
    const bf16* __restrict__ hbf,   // [BN][128]
    const float* __restrict__ hold, // [BN][128]
    const bf16* __restrict__ Wt,    // [384][DIN]
    const bf16* __restrict__ Ut,    // [384][128]
    const float* __restrict__ bias, // [2][384]
    float* __restrict__ hnew) {
    constexpr int XS = DIN + 8;
    constexpr int HS = 136;
    constexpr int GS = G3H + 8;
    __shared__ __align__(16) bf16 Ax[16 * XS];
    __shared__ __align__(16) bf16 Ah[16 * HS];
    __shared__ float Gb[16 * GS];
    __shared__ float Hb[16 * HS];

    int tid     = threadIdx.x;
    int rowbase = blockIdx.x * 16;

    constexpr int XCH = DIN / 8;  // 16B chunks per x row
    for (int i = tid; i < 16 * XCH; i += 768) {
        int r = i / XCH, c = (i - r * XCH) * 8;
        async_copy16(&xbf[(long)(rowbase + r) * DIN + c], &Ax[r * XS + c]);
    }
    for (int i = tid; i < 16 * 16; i += 768) {
        int r = i >> 4, c = (i & 15) * 8;
        async_copy16(&hbf[(long)(rowbase + r) * Hc + c], &Ah[r * HS + c]);
    }
    async_wait();
    __syncthreads();

    int w    = tid >> 5;   // 0..23 -> gate col tile
    int lane = tid & 31;
    int nloc = w * 16 + (lane & 15);
    int half = lane >> 4;
    int mrow16 = lane & 15;

    const bf16* Wrow = Wt + (long)nloc * DIN;
    const bf16* Urow = Ut + (long)nloc * Hc;
    __builtin_prefetch(Wrow, 0, 3);
    __builtin_prefetch(Urow, 0, 3);

    floatx8 accX = {0.f, 0.f, 0.f, 0.f, 0.f, 0.f, 0.f, 0.f};
    floatx8 accH = {0.f, 0.f, 0.f, 0.f, 0.f, 0.f, 0.f, 0.f};
#pragma unroll
    for (int kb = 0; kb < DIN; kb += 32) {
        V16 a, b;
        const bf16* arow = &Ax[mrow16 * XS + kb + 8 * half];
        a.q[0] = *(const uint4*)(arow);
        a.q[1] = *(const uint4*)(arow + 16);
        const bf16* brow = Wrow + kb + 8 * half;
        b.q[0] = *(const uint4*)(brow);
        b.q[1] = *(const uint4*)(brow + 16);
        accX = wmma_bf16(a.v, b.v, accX);
    }
#pragma unroll
    for (int kb = 0; kb < Hc; kb += 32) {
        V16 a, b;
        const bf16* arow = &Ah[mrow16 * HS + kb + 8 * half];
        a.q[0] = *(const uint4*)(arow);
        a.q[1] = *(const uint4*)(arow + 16);
        const bf16* brow = Urow + kb + 8 * half;
        b.q[0] = *(const uint4*)(brow);
        b.q[1] = *(const uint4*)(brow + 16);
        accH = wmma_bf16(a.v, b.v, accH);
    }

    float b0 = bias[nloc], b1 = bias[G3H + nloc];
#pragma unroll
    for (int v = 0; v < 8; ++v) {
        int mr = v + 8 * half;
        if (w < 16) {  // z / r columns: xg + hg combined
            Gb[mr * GS + nloc] = accX[v] + accH[v] + b0 + b1;
        } else {       // candidate columns: keep xh and hh separate
            Gb[mr * GS + nloc]          = accX[v] + b0;
            Hb[mr * HS + (nloc - 256)]  = accH[v] + b1;
        }
    }
    __syncthreads();

    for (int i = tid; i < 16 * Hc; i += 768) {
        int mr = i >> 7;
        int j  = i & 127;
        float z    = 1.f / (1.f + __expf(-Gb[mr * GS + j]));
        float r    = 1.f / (1.f + __expf(-Gb[mr * GS + 128 + j]));
        float cand = tanhf(Gb[mr * GS + 256 + j] + r * Hb[mr * HS + j]);
        float ho   = hold[(long)(rowbase + mr) * Hc + j];
        hnew[(long)(rowbase + mr) * Hc + j] = z * ho + (1.f - z) * cand;
    }
}

// ------------------------------- launcher ----------------------------------

extern "C" void kernel_launch(void* const* d_in, const int* in_sizes, int n_in,
                              void* d_out, int out_size, void* d_ws, size_t ws_size,
                              hipStream_t stream) {
    (void)in_sizes; (void)n_in; (void)out_size; (void)ws_size;

    const float* states = (const float*)d_in[0];
    const int*   eids   = (const int*)d_in[1];
    const float* typeW  = (const float*)d_in[2];
    const float* typeB  = (const float*)d_in[3];
    const float* gW[4], *gU[4], *gb[4];
    const int din_l[4] = {128, 256, 128, 384};
    for (int l = 0; l < 4; ++l) {
        gW[l] = (const float*)d_in[4 + 3 * l];
        gU[l] = (const float*)d_in[5 + 3 * l];
        gb[l] = (const float*)d_in[6 + 3 * l];
    }

    // workspace carve-out (~53 MB total)
    char* wsb = (char*)d_ws;
    size_t off = 0;
    auto carve = [&](size_t bytes) -> void* {
        off = (off + 255) & ~(size_t)255;
        void* p = wsb + off;
        off += bytes;
        return p;
    };
    Meta*  meta   = (Meta*)carve(sizeof(Meta));
    int*   order  = (int*)carve((size_t)Ec * 4);
    float* hA     = (float*)carve((size_t)BNc * Hc * 4);
    float* hB     = (float*)carve((size_t)BNc * Hc * 4);
    float* L0out  = (float*)carve((size_t)BNc * Hc * 4);
    float* agg    = (float*)carve((size_t)BNc * Hc * 4);
    bf16*  WtType = (bf16*)carve((size_t)4 * Tc * Hc * Hc * 2);
    bf16*  WtG[4], *UtG[4];
    for (int l = 0; l < 4; ++l) {
        WtG[l] = (bf16*)carve((size_t)din_l[l] * G3H * 2);
        UtG[l] = (bf16*)carve((size_t)Hc * G3H * 2);
    }
    bf16* xbf = (bf16*)carve((size_t)BNc * G3H * 2);
    bf16* hbf = (bf16*)carve((size_t)BNc * Hc * 2);

    // ---- per-launch prep: bucket edges by type, bf16-transpose weights ----
    (void)hipMemsetAsync(meta, 0, sizeof(Meta), stream);
    k_hist<<<(Ec + 255) / 256, 256, 0, stream>>>(eids, meta);
    k_scan<<<1, 1, 0, stream>>>(meta);
    k_scatter<<<(Ec + 255) / 256, 256, 0, stream>>>(eids, meta, order);

    {
        long n = (long)4 * Tc * Hc * Hc;
        k_transpose_bf16<<<(int)((n + 255) / 256), 256, 0, stream>>>(WtType, typeW, 4 * Tc, Hc, Hc);
    }
    for (int l = 0; l < 4; ++l) {
        long nw = (long)din_l[l] * G3H;
        k_transpose_bf16<<<(int)((nw + 255) / 256), 256, 0, stream>>>(WtG[l], gW[l], 1, din_l[l], G3H);
        long nu = (long)Hc * G3H;
        k_transpose_bf16<<<(int)((nu + 255) / 256), 256, 0, stream>>>(UtG[l], gU[l], 1, Hc, G3H);
    }
    k_copy_f32<<<(BNc * Hc + 255) / 256, 256, 0, stream>>>(hA, states, BNc * Hc);

    // ---- 8 propagate steps: L0 x3, L1 x1, L2 x3, L3 x1 ----
    float* out = (float*)d_out;
    struct Step { int layer; const float* hin; float* hout; };
    const Step steps[8] = {
        {0, hA, hB}, {0, hB, hA}, {0, hA, L0out},
        {1, L0out, hA},
        {2, hA, hB}, {2, hB, hA}, {2, hA, hB},
        {3, hB, out},
    };

    const int maxTiles = Ec / 64 + Tc;  // upper bound on per-type padded tiles
    for (int s = 0; s < 8; ++s) {
        int l = steps[s].layer;
        const float* hin = steps[s].hin;
        float* hout      = steps[s].hout;

        // bf16 copy of current state (edge-gather source AND GRU h operand)
        k_f32_to_bf16<<<(BNc * Hc + 255) / 256, 256, 0, stream>>>(hbf, hin, BNc * Hc);

        (void)hipMemsetAsync(agg, 0, (size_t)BNc * Hc * 4, stream);
        k_edge_gemm<<<maxTiles, 1024, 0, stream>>>(hbf, eids, order, meta, WtType,
                                                   typeB, agg, l);

        const float* p0; const float* p1; const float* p2;
        int n0, n1, n2;
        switch (l) {
            case 1:  p0 = states; n0 = 128; p1 = agg;   n1 = 128; p2 = agg; n2 = 0;   break;
            case 3:  p0 = states; n0 = 128; p1 = L0out; n1 = 128; p2 = agg; n2 = 128; break;
            default: p0 = agg;    n0 = 128; p1 = agg;   n1 = 0;   p2 = agg; n2 = 0;   break;
        }
        int din = n0 + n1 + n2;
        {
            long n = (long)BNc * din;
            k_concat_bf16<<<(int)((n + 255) / 256), 256, 0, stream>>>(xbf, p0, p1, p2, n0, n1, n2);
        }

        if (din == 128)
            k_gru<128><<<BNc / 16, 768, 0, stream>>>(xbf, hbf, hin, WtG[l], UtG[l], gb[l], hout);
        else if (din == 256)
            k_gru<256><<<BNc / 16, 768, 0, stream>>>(xbf, hbf, hin, WtG[l], UtG[l], gb[l], hout);
        else
            k_gru<384><<<BNc / 16, 768, 0, stream>>>(xbf, hbf, hin, WtG[l], UtG[l], gb[l], hout);
    }
}
